// CLIP_80616536146746
// MI455X (gfx1250) — matrix operand order repaired and checked by reading, not compiled, not run on hardware
//
#include <hip/hip_runtime.h>
#include <hip/hip_bf16.h>
#include <math.h>

// ---------------------------------------------------------------------------
// CLIP forward on MI455X (gfx1250): all big GEMMs via v_wmma_f32_16x16x32_f16
// LDS tiles stored in WMMA fragment order -> ds_load_b128 fragment loads.
// Wave tile 32x64 (8 WMMA / 12 fragment loads per K-step).
// ---------------------------------------------------------------------------

typedef __attribute__((ext_vector_type(16))) _Float16 v16h;
typedef __attribute__((ext_vector_type(8)))  _Float16 v8h;
typedef __attribute__((ext_vector_type(8)))  float    v8f;

#define TBM 64
#define TBN 256
#define TBK 32
#define AROW 40   // f16 elements per A row (32 data + 8 pad), 80B: 16B aligned chunks
#define BROW 40   // f16 elements per B row (32 data + 8 pad)

__device__ __forceinline__ float gelu_exact(float x) {
  return 0.5f * x * (1.0f + erff(x * 0.70710678118654752f));
}

__device__ __forceinline__ v16h ld_frag(const _Float16* p) {
  v8h lo = *(const v8h*)p;         // ds_load_b128
  v8h hi = *(const v8h*)(p + 8);   // ds_load_b128
  return __builtin_shufflevector(lo, hi, 0, 1, 2, 3, 4, 5, 6, 7,
                                 8, 9, 10, 11, 12, 13, 14, 15);
}

__device__ __forceinline__ void store_tile(v8f acc, int m0, int n0, int hf, int l16,
                                           const float* bias, const float* res,
                                           float* C, int ldc, int M, int N, int act) {
  int n = n0 + l16;
#pragma unroll
  for (int vv = 0; vv < 8; ++vv) {
    int m = m0 + hf * 8 + vv;
    if (m < M && n < N) {
      float val = acc[vv];
      if (bias) val += bias[n];
      if (act)  val  = gelu_exact(val);
      if (res)  val += res[(size_t)m * ldc + n];
      C[(size_t)m * ldc + n] = val;
    }
  }
}

// C[M,N] = act(A[M,K] @ B[K,N] + bias) + residual   (f16 WMMA, f32 accumulate)
// REQUIREMENT (all call sites satisfy): K % 32 == 0.
__global__ __launch_bounds__(256) void gemm_wmma_kernel(
    const float* __restrict__ A, int lda,
    const float* __restrict__ Bm, int ldb,
    const float* __restrict__ bias,
    const float* __restrict__ res,
    float* __restrict__ C, int ldc,
    int M, int N, int K, int act) {
  // A stored per-row contiguous: [k0..7 | k8..15 | k16..23 | k24..31]
  // -> lane half hf reads contiguous 16 f16 at element offset hf*16.
  // B stored transposed [n][k]: B frag uses k = hf*16+e at fixed n, same order as A
  // -> contiguous 16 f16 at element offset hf*16.
  __shared__ _Float16 As[TBM * AROW];   //  5.0 KB
  __shared__ _Float16 Bs[TBN * BROW];   // 20.0 KB

  const int tid  = threadIdx.x;
  const int bm0  = blockIdx.y * TBM;
  const int bn0  = blockIdx.x * TBN;
  const int w    = tid >> 5;
  const int lane = tid & 31;
  const int wm   = w >> 2;      // 0..1 : M sub-tile of 32
  const int wn   = w & 3;       // 0..3 : N sub-tile of 64
  const int hf   = lane >> 4;   // 0..1
  const int l16  = lane & 15;

  // staging coordinates (fixed per thread)
  const int am = tid >> 2;              // A row 0..63
  const int ag = tid & 3;               // A k-group of 8
  const int adoff = ag * 8;  // contiguous dest
  const int bnl = tid;                  // B column 0..255

  // fragment base pointers (fixed per thread)
  const _Float16* a0p = &As[(wm * 32 + l16) * AROW + hf * 16];
  const _Float16* a1p = a0p + 16 * AROW;
  const _Float16* b0p = &Bs[(wn * 64 + l16) * BROW + hf * 16];

  v8f c00 = {}, c01 = {}, c02 = {}, c03 = {};
  v8f c10 = {}, c11 = {}, c12 = {}, c13 = {};

  for (int k0 = 0; k0 < K; k0 += TBK) {
    // ---- stage A tile (64 x 32): one v8h per thread, vector global read ----
    {
      const int gr = bm0 + am;
      v8h pk;
      if (gr < M) {
        const float* ap = A + (size_t)gr * lda + k0 + ag * 8;
#pragma unroll
        for (int i = 0; i < 8; ++i) pk[i] = (_Float16)ap[i];
        if (k0 + TBK < K) __builtin_prefetch(ap + TBK, 0, 0);
      } else {
#pragma unroll
        for (int i = 0; i < 8; ++i) pk[i] = (_Float16)0.0f;
      }
      *(v8h*)&As[am * AROW + adoff] = pk;   // ds_store_b128
    }
    // ---- stage B tile transposed (256 n x 32 k): four v8h per thread ----
    {
      const int gn = bn0 + bnl;
      v8h p0, p1, p2, p3;
      if (gn < N) {
        const float* bp = Bm + (size_t)k0 * ldb + gn;
#pragma unroll
        for (int i = 0; i < 8; ++i) p0[i] = (_Float16)bp[(size_t)i * ldb];
#pragma unroll
        for (int i = 0; i < 8; ++i) p1[i] = (_Float16)bp[(size_t)(i + 8) * ldb];
#pragma unroll
        for (int i = 0; i < 8; ++i) p2[i] = (_Float16)bp[(size_t)(i + 16) * ldb];
#pragma unroll
        for (int i = 0; i < 8; ++i) p3[i] = (_Float16)bp[(size_t)(i + 24) * ldb];
        if (k0 + TBK < K) __builtin_prefetch(bp + (size_t)TBK * ldb, 0, 0);
      } else {
#pragma unroll
        for (int i = 0; i < 8; ++i) {
          p0[i] = (_Float16)0.0f; p1[i] = (_Float16)0.0f;
          p2[i] = (_Float16)0.0f; p3[i] = (_Float16)0.0f;
        }
      }
      _Float16* br = &Bs[bnl * BROW];
      *(v8h*)(br + 0)  = p0;  // ds_store_b128
      *(v8h*)(br + 8)  = p1;  // ds_store_b128
      *(v8h*)(br + 16) = p2;  // ds_store_b128
      *(v8h*)(br + 24) = p3;  // ds_store_b128
    }
    __syncthreads();

    v16h A0 = ld_frag(a0p);
    v16h A1 = ld_frag(a1p);
    v16h B0 = ld_frag(b0p);
    v16h B1 = ld_frag(b0p + 16 * BROW);
    v16h B2 = ld_frag(b0p + 32 * BROW);
    v16h B3 = ld_frag(b0p + 48 * BROW);

    c00 = __builtin_amdgcn_wmma_f32_16x16x32_f16(false, A0, false, B0, (short)0, c00, false, false);
    c01 = __builtin_amdgcn_wmma_f32_16x16x32_f16(false, A0, false, B1, (short)0, c01, false, false);
    c02 = __builtin_amdgcn_wmma_f32_16x16x32_f16(false, A0, false, B2, (short)0, c02, false, false);
    c03 = __builtin_amdgcn_wmma_f32_16x16x32_f16(false, A0, false, B3, (short)0, c03, false, false);
    c10 = __builtin_amdgcn_wmma_f32_16x16x32_f16(false, A1, false, B0, (short)0, c10, false, false);
    c11 = __builtin_amdgcn_wmma_f32_16x16x32_f16(false, A1, false, B1, (short)0, c11, false, false);
    c12 = __builtin_amdgcn_wmma_f32_16x16x32_f16(false, A1, false, B2, (short)0, c12, false, false);
    c13 = __builtin_amdgcn_wmma_f32_16x16x32_f16(false, A1, false, B3, (short)0, c13, false, false);
    __syncthreads();
  }

  const int mBase = bm0 + wm * 32;
  const int nBase = bn0 + wn * 64;
  store_tile(c00, mBase + 0,  nBase + 0,  hf, l16, bias, res, C, ldc, M, N, act);
  store_tile(c01, mBase + 0,  nBase + 16, hf, l16, bias, res, C, ldc, M, N, act);
  store_tile(c02, mBase + 0,  nBase + 32, hf, l16, bias, res, C, ldc, M, N, act);
  store_tile(c03, mBase + 0,  nBase + 48, hf, l16, bias, res, C, ldc, M, N, act);
  store_tile(c10, mBase + 16, nBase + 0,  hf, l16, bias, res, C, ldc, M, N, act);
  store_tile(c11, mBase + 16, nBase + 16, hf, l16, bias, res, C, ldc, M, N, act);
  store_tile(c12, mBase + 16, nBase + 32, hf, l16, bias, res, C, ldc, M, N, act);
  store_tile(c13, mBase + 16, nBase + 48, hf, l16, bias, res, C, ldc, M, N, act);
}

// ---------------------------------------------------------------------------
// Helper kernels (VALU; tiny fraction of total work)
// ---------------------------------------------------------------------------

__device__ __forceinline__ float pe_val(int pos, int i, int d) {
  float ie  = (float)(i & ~1);
  float ang = (float)pos * powf(10000.0f, -ie / (float)d);
  return (i & 1) ? cosf(ang) : sinf(ang);
}

// patches[b*196+p, c*256+py*16+px] = image[b, c, gy*16+py, gx*16+px]
__global__ __launch_bounds__(256) void build_patches_kernel(const float* __restrict__ img,
                                                            float* __restrict__ patches) {
  int idx = blockIdx.x * 256 + threadIdx.x;
  const int total = 32 * 196 * 768;
  if (idx >= total) return;
  int d  = idx % 768;
  int rp = idx / 768;
  int p  = rp % 196;
  int b  = rp / 196;
  int c   = d >> 8;
  int rem = d & 255;
  int py  = rem >> 4;
  int px  = rem & 15;
  int gy = p / 14, gx = p % 14;
  patches[idx] = img[((size_t)(b * 3 + c) * 224 + (gy * 16 + py)) * 224 + gx * 16 + px];
}

__global__ __launch_bounds__(256) void assemble_vit_kernel(const float* __restrict__ pemb,
                                                           const float* __restrict__ cls,
                                                           float* __restrict__ x) {
  int idx = blockIdx.x * 256 + threadIdx.x;
  const int total = 32 * 197 * 768;
  if (idx >= total) return;
  int d  = idx % 768;
  int rs = idx / 768;
  int s  = rs % 197;
  int b  = rs / 197;
  float base = (s == 0) ? cls[d] : pemb[((size_t)b * 196 + (s - 1)) * 768 + d];
  x[idx] = base + pe_val(s, d, 768);
}

__global__ __launch_bounds__(256) void assemble_txt_kernel(const int* __restrict__ text,
                                                           const float* __restrict__ emb,
                                                           const float* __restrict__ cls,
                                                           float* __restrict__ x) {
  int idx = blockIdx.x * 256 + threadIdx.x;
  const int total = 32 * 77 * 512;
  if (idx >= total) return;
  int d  = idx % 512;
  int rs = idx / 512;
  int s  = rs % 77;
  int b  = rs / 77;
  float base;
  if (s == 0) base = cls[d];
  else {
    int tok = text[b * 76 + (s - 1)];
    base = emb[(size_t)tok * 512 + d];
  }
  x[idx] = base + pe_val(s, d, 512);
}

__global__ __launch_bounds__(256) void ln_kernel(const float* __restrict__ x,
                                                 const float* __restrict__ g,
                                                 const float* __restrict__ bb,
                                                 float* __restrict__ out, int D) {
  __shared__ float red[256];
  int row = blockIdx.x, tid = threadIdx.x;
  const float* xr = x + (size_t)row * D;
  float s = 0.0f;
  for (int i = tid; i < D; i += 256) s += xr[i];
  red[tid] = s; __syncthreads();
  for (int st = 128; st > 0; st >>= 1) { if (tid < st) red[tid] += red[tid + st]; __syncthreads(); }
  float mu = red[0] / (float)D; __syncthreads();
  float vs = 0.0f;
  for (int i = tid; i < D; i += 256) { float dd = xr[i] - mu; vs += dd * dd; }
  red[tid] = vs; __syncthreads();
  for (int st = 128; st > 0; st >>= 1) { if (tid < st) red[tid] += red[tid + st]; __syncthreads(); }
  float inv = rsqrtf(red[0] / (float)D + 1e-5f);
  float* orow = out + (size_t)row * D;
  for (int i = tid; i < D; i += 256) orow[i] = (xr[i] - mu) * inv * g[i] + bb[i];
}

// one block per (q_index, head, batch): scores -> softmax -> weighted V sum
__global__ __launch_bounds__(128) void attn_kernel(const float* __restrict__ q,
                                                   const float* __restrict__ k,
                                                   const float* __restrict__ v,
                                                   float* __restrict__ o,
                                                   const int* __restrict__ mask,
                                                   int S, int H, int dk, float scale) {
  __shared__ float sc[224];
  __shared__ float red[128];
  int b = blockIdx.z, h = blockIdx.y, qi = blockIdx.x;
  int D = H * dk;
  int tid = threadIdx.x;
  const float* qr = q + ((size_t)(b * S + qi) * D + (size_t)h * dk);
  for (int j = tid; j < S; j += 128) {
    const float* kr = k + ((size_t)(b * S + j) * D + (size_t)h * dk);
    float s = 0.0f;
    for (int d = 0; d < dk; ++d) s += qr[d] * kr[d];
    s *= scale;
    if (mask && mask[(size_t)b * S * S + j] == 0) s = -1e30f;
    sc[j] = s;
  }
  __syncthreads();
  float m = -1e30f;
  for (int j = tid; j < S; j += 128) m = fmaxf(m, sc[j]);
  red[tid] = m; __syncthreads();
  for (int st = 64; st > 0; st >>= 1) { if (tid < st) red[tid] = fmaxf(red[tid], red[tid + st]); __syncthreads(); }
  float mx = red[0]; __syncthreads();
  float sum = 0.0f;
  for (int j = tid; j < S; j += 128) { float e = __expf(sc[j] - mx); sc[j] = e; sum += e; }
  red[tid] = sum; __syncthreads();
  for (int st = 64; st > 0; st >>= 1) { if (tid < st) red[tid] += red[tid + st]; __syncthreads(); }
  float inv = 1.0f / red[0];
  __syncthreads();
  if (tid < dk) {
    float acc = 0.0f;
    for (int j = 0; j < S; ++j)
      acc += sc[j] * v[((size_t)(b * S + j) * D + (size_t)h * dk) + tid];
    o[((size_t)(b * S + qi) * D + (size_t)h * dk) + tid] = acc * inv;
  }
}

__global__ __launch_bounds__(256) void pool_text_kernel(const float* __restrict__ x,
                                                        const int* __restrict__ mask,
                                                        float* __restrict__ pooled,
                                                        int S, int D) {
  __shared__ int lastIdx;
  int b = blockIdx.x;
  if (threadIdx.x == 0) {
    int cnt = 0;
    for (int j = 0; j < S; ++j) cnt += mask[(size_t)b * S * S + j];
    lastIdx = cnt - 1;
  }
  __syncthreads();
  for (int d = threadIdx.x; d < D; d += 256)
    pooled[(size_t)b * D + d] = x[((size_t)b * S + lastIdx) * D + d];
}

__global__ __launch_bounds__(256) void l2_kernel(float* __restrict__ x, int D) {
  __shared__ float red[256];
  int row = blockIdx.x, tid = threadIdx.x;
  float* xr = x + (size_t)row * D;
  float s = 0.0f;
  for (int i = tid; i < D; i += 256) s += xr[i] * xr[i];
  red[tid] = s; __syncthreads();
  for (int st = 128; st > 0; st >>= 1) { if (tid < st) red[tid] += red[tid + st]; __syncthreads(); }
  float inv = rsqrtf(red[0]);
  for (int i = tid; i < D; i += 256) xr[i] *= inv;
}

__global__ __launch_bounds__(1024) void loss_kernel(const float* __restrict__ Ie,
                                                    const float* __restrict__ Te,
                                                    const float* __restrict__ temp,
                                                    float* __restrict__ out) {
  __shared__ float lg[32 * 32];
  __shared__ float rowl[32], coll[32];
  int tid = threadIdx.x;
  float scale = __expf(temp[0]);
  int i = tid >> 5, j = tid & 31;
  float s = 0.0f;
  for (int d = 0; d < 512; ++d) s += Ie[i * 512 + d] * Te[j * 512 + d];
  lg[i * 32 + j] = s * scale;
  __syncthreads();
  if (tid < 32) {
    float mx = -1e30f;
    for (int j2 = 0; j2 < 32; ++j2) mx = fmaxf(mx, lg[tid * 32 + j2]);
    float sm = 0.0f;
    for (int j2 = 0; j2 < 32; ++j2) sm += __expf(lg[tid * 32 + j2] - mx);
    rowl[tid] = mx + __logf(sm);
  } else if (tid < 64) {
    int j2 = tid - 32;
    float mx = -1e30f;
    for (int i2 = 0; i2 < 32; ++i2) mx = fmaxf(mx, lg[i2 * 32 + j2]);
    float sm = 0.0f;
    for (int i2 = 0; i2 < 32; ++i2) sm += __expf(lg[i2 * 32 + j2] - mx);
    coll[j2] = mx + __logf(sm);
  }
  __syncthreads();
  if (tid == 0) {
    float lt = 0.0f, li = 0.0f;
    for (int d = 0; d < 32; ++d) {
      lt += rowl[d] - lg[d * 32 + d];
      li += coll[d] - lg[d * 32 + d];
    }
    out[0] = 0.5f * (lt + li) / 32.0f;
  }
}

// ---------------------------------------------------------------------------
// Host orchestration
// ---------------------------------------------------------------------------

static void gemm(hipStream_t s, const float* A, int lda, const float* B, int ldb,
                 const float* bias, const float* res, float* C, int ldc,
                 int M, int N, int K, int act) {
  dim3 grid((N + TBN - 1) / TBN, (M + TBM - 1) / TBM);
  gemm_wmma_kernel<<<grid, dim3(256), 0, s>>>(A, lda, B, ldb, bias, res, C, ldc, M, N, K, act);
}

struct LayerP {
  const float *ln1g, *ln1b, *Wq, *bq, *Wk, *bk, *Wv, *bv, *Wo, *bo, *ln2g, *ln2b, *W1, *b1, *W2, *b2;
};

extern "C" void kernel_launch(void* const* d_in, const int* in_sizes, int n_in,
                              void* d_out, int out_size, void* d_ws, size_t ws_size,
                              hipStream_t stream) {
  (void)in_sizes; (void)n_in; (void)out_size; (void)ws_size;

  const float* image = (const float*)d_in[0];
  const int*   text  = (const int*)d_in[1];
  const int*   maskp = (const int*)d_in[2];

  int p = 3;
  auto F = [&](int i) { return (const float*)d_in[i]; };
  const float* patch_W = F(p++);
  const float* patch_b = F(p++);
  const float* vit_cls = F(p++);
  LayerP vl[6];
  for (int l = 0; l < 6; ++l) {
    vl[l].ln1g = F(p++); vl[l].ln1b = F(p++);
    vl[l].Wq = F(p++); vl[l].bq = F(p++);
    vl[l].Wk = F(p++); vl[l].bk = F(p++);
    vl[l].Wv = F(p++); vl[l].bv = F(p++);
    vl[l].Wo = F(p++); vl[l].bo = F(p++);
    vl[l].ln2g = F(p++); vl[l].ln2b = F(p++);
    vl[l].W1 = F(p++); vl[l].b1 = F(p++);
    vl[l].W2 = F(p++); vl[l].b2 = F(p++);
  }
  const float* vit_proj = F(p++);
  const float* txt_emb  = F(p++);
  const float* txt_cls  = F(p++);
  LayerP tl[6];
  for (int l = 0; l < 6; ++l) {
    tl[l].ln1g = F(p++); tl[l].ln1b = F(p++);
    tl[l].Wq = F(p++); tl[l].bq = F(p++);
    tl[l].Wk = F(p++); tl[l].bk = F(p++);
    tl[l].Wv = F(p++); tl[l].bv = F(p++);
    tl[l].Wo = F(p++); tl[l].bo = F(p++);
    tl[l].ln2g = F(p++); tl[l].ln2b = F(p++);
    tl[l].W1 = F(p++); tl[l].b1 = F(p++);
    tl[l].W2 = F(p++); tl[l].b2 = F(p++);
  }
  const float* txt_proj = F(p++);
  const float* temp     = F(p++);

  // ----- workspace layout (floats) -----
  const size_t QS = (size_t)6304 * 768;   // 4,841,472
  float* ws  = (float*)d_ws;
  float* X   = ws;                 // activations  (<= QS)
  float* Hh  = ws + QS;            // LN output    (<= QS)
  float* BIG = ws + 2 * QS;        // q/k/v/o or mid (4*QS)
  float* SM  = ws + 6 * QS;        // pooled / embeds
  float* Qb = BIG, *Kb = BIG + QS, *Vb = BIG + 2 * QS, *Ob = BIG + 3 * QS;
  float* IeRaw = SM;               // [32,512]
  float* TeRaw = SM + 32 * 512;    // [32,512]
  float* pooledT = SM + 2 * 32 * 512; // [32,512]

  const int MV = 32 * 197, MT = 32 * 77;

  // ===================== image encoder =====================
  {
    const int totP = 32 * 196 * 768;
    build_patches_kernel<<<(totP + 255) / 256, 256, 0, stream>>>(image, BIG);      // patches
    gemm(stream, BIG, 768, patch_W, 768, patch_b, nullptr, BIG + QS, 768, 32 * 196, 768, 768, 0);
    const int totX = MV * 768;
    assemble_vit_kernel<<<(totX + 255) / 256, 256, 0, stream>>>(BIG + QS, vit_cls, X);
  }
  for (int l = 0; l < 6; ++l) {
    const LayerP& L = vl[l];
    ln_kernel<<<MV, 256, 0, stream>>>(X, L.ln1g, L.ln1b, Hh, 768);
    gemm(stream, Hh, 768, L.Wq, 768, L.bq, nullptr, Qb, 768, MV, 768, 768, 0);
    gemm(stream, Hh, 768, L.Wk, 768, L.bk, nullptr, Kb, 768, MV, 768, 768, 0);
    gemm(stream, Hh, 768, L.Wv, 768, L.bv, nullptr, Vb, 768, MV, 768, 768, 0);
    attn_kernel<<<dim3(197, 12, 32), 128, 0, stream>>>(Qb, Kb, Vb, Ob, nullptr, 197, 12, 64, 0.125f);
    gemm(stream, Ob, 768, L.Wo, 768, L.bo, X, X, 768, MV, 768, 768, 0);
    ln_kernel<<<MV, 256, 0, stream>>>(X, L.ln2g, L.ln2b, Hh, 768);
    gemm(stream, Hh, 768, L.W1, 3072, L.b1, nullptr, BIG, 3072, MV, 3072, 768, 1);  // GELU
    gemm(stream, BIG, 3072, L.W2, 768, L.b2, X, X, 768, MV, 768, 3072, 0);
  }
  // CLS pooling folded into GEMM via lda = 197*768
  gemm(stream, X, 197 * 768, vit_proj, 512, nullptr, nullptr, IeRaw, 512, 32, 512, 768, 0);
  l2_kernel<<<32, 256, 0, stream>>>(IeRaw, 512);

  // ===================== text encoder =====================
  {
    const int totX = MT * 512;
    assemble_txt_kernel<<<(totX + 255) / 256, 256, 0, stream>>>(text, txt_emb, txt_cls, X);
  }
  for (int l = 0; l < 6; ++l) {
    const LayerP& L = tl[l];
    ln_kernel<<<MT, 256, 0, stream>>>(X, L.ln1g, L.ln1b, Hh, 512);
    gemm(stream, Hh, 512, L.Wq, 512, L.bq, nullptr, Qb, 512, MT, 512, 512, 0);
    gemm(stream, Hh, 512, L.Wk, 512, L.bk, nullptr, Kb, 512, MT, 512, 512, 0);
    gemm(stream, Hh, 512, L.Wv, 512, L.bv, nullptr, Vb, 512, MT, 512, 512, 0);
    attn_kernel<<<dim3(77, 8, 32), 128, 0, stream>>>(Qb, Kb, Vb, Ob, maskp, 77, 8, 64, 0.125f);
    gemm(stream, Ob, 512, L.Wo, 512, L.bo, X, X, 512, MT, 512, 512, 0);
    ln_kernel<<<MT, 256, 0, stream>>>(X, L.ln2g, L.ln2b, Hh, 512);
    gemm(stream, Hh, 512, L.W1, 2048, L.b1, nullptr, BIG, 2048, MT, 2048, 512, 1);  // GELU
    gemm(stream, BIG, 2048, L.W2, 512, L.b2, X, X, 512, MT, 512, 2048, 0);
  }
  pool_text_kernel<<<32, 256, 0, stream>>>(X, maskp, pooledT, 77, 512);
  gemm(stream, pooledT, 512, txt_proj, 512, nullptr, nullptr, TeRaw, 512, 32, 512, 512, 0);
  l2_kernel<<<32, 256, 0, stream>>>(TeRaw, 512);

  // ===================== contrastive loss =====================
  loss_kernel<<<1, 1024, 0, stream>>>(IeRaw, TeRaw, temp, (float*)d_out);
}
